// HPN_65369402245667
// MI455X (gfx1250) — compile-verified, matrix operand order checked by reading, not run on hardware
//
#include <hip/hip_runtime.h>
#include <hip/hip_bf16.h>

typedef __attribute__((ext_vector_type(16))) _Float16 v16h;
typedef __attribute__((ext_vector_type(8)))  float    v8f;

#define Dm    128
#define S_LEN 1000
#define B_SZ  512
#define M_BIG (B_SZ * S_LEN)   // 512000
#define MT    96               // rows per block (6 waves x 16 rows)
#define BLK   192              // 6 wave32 waves
#define NWAVE 6

// Fragment-major LDS sizes (halfs)
#define SAF_HALFS (NWAVE * 4 * 32 * 16)   // 12288 halfs = 24 KB
#define SBF_HALFS (4 * 8 * 32 * 16)       // 16384 halfs = 32 KB

union PackH { unsigned int u; _Float16 h[2]; };

__device__ __forceinline__ unsigned int pack2(float a, float b) {
  PackH p; p.h[0] = (_Float16)a; p.h[1] = (_Float16)b; return p.u;
}

// K (0..127) -> (kt = K-tile, j = fragment element, half = lane group)
__device__ __forceinline__ void kmap(int k, int& kt, int& j, int& half) {
  kt   = k >> 5;
  int kk = k & 31;
  half = (kk >> 3) & 1;
  j    = (((kk >> 4) & 1) << 3) + (kk & 7);
}

// ---- staging: fp32 global -> f16 LDS in fragment-major (swizzled) layout --
__device__ __forceinline__ void stage_A_frag(_Float16* sAf, const float* __restrict__ A,
                                             long row0, int M, int t) {
  for (int i = t; i < MT * 64; i += BLK) {
    int r = i >> 6, k = (i & 63) << 1;       // row in tile, even K
    long gr = row0 + r;
    float2 v = make_float2(0.f, 0.f);
    if (gr < (long)M) v = *(const float2*)(A + gr * Dm + k);
    int kt, j, half; kmap(k, kt, j, half);
    int lane = half * 16 + (r & 15);
    int w = r >> 4;
    int idx = ((((w * 4 + kt) * 32 + lane) * 16) + j) >> 1;
    ((unsigned int*)sAf)[idx] = pack2(v.x, v.y);
  }
}

// W row-major [N][K]; fragment element j of lane (half*16 + n%16), tile (kt, nt)
__device__ __forceinline__ void stage_B_frag(_Float16* sBf, const float* __restrict__ W, int t) {
  for (int i = t; i < Dm * 64; i += BLK) {
    int n = i >> 6, k = (i & 63) << 1;
    float2 v = *(const float2*)(W + n * Dm + k);
    int kt, j, half; kmap(k, kt, j, half);
    int lane = half * 16 + (n & 15);
    int nt = n >> 4;
    int idx = ((((kt * 8 + nt) * 32 + lane) * 16) + j) >> 1;
    ((unsigned int*)sBf)[idx] = pack2(v.x, v.y);
  }
}

// ---- wave-level GEMM: 16 rows x 128 cols, K=128, f32 accumulate -----------
// Fragment loads are 32 contiguous bytes per lane -> 2x ds_load_b128 each.
__device__ __forceinline__ void wave_gemm(const _Float16* sAf, const _Float16* sBf,
                                          int wave, int lane, v8f acc[8]) {
#pragma unroll
  for (int kt = 0; kt < 4; ++kt) {
    v16h a = *(const v16h*)(sAf + ((wave * 4 + kt) * 32 + lane) * 16);
#pragma unroll
    for (int nt = 0; nt < 8; ++nt) {
      v16h b = *(const v16h*)(sBf + ((kt * 8 + nt) * 32 + lane) * 16);
      acc[nt] = __builtin_amdgcn_wmma_f32_16x16x32_f16(
          false, a, false, b, (short)0, acc[nt], false, false);
    }
  }
}

// ---- fused layer: out = r*(ctx@W^T + bW) + (1-r)*relu(scale*(ctx@A^T) + bA)
__global__ __launch_bounds__(BLK) void layer_kernel(
    const float* __restrict__ ctx_in, float* __restrict__ ctx_out,
    const float* __restrict__ Ww, const float* __restrict__ Wb,
    const float* __restrict__ Aw, const float* __restrict__ Ab,
    const float* __restrict__ r_all, int ridx, int M, float scale) {
  __shared__ alignas(32) _Float16 sAf[SAF_HALFS];
  __shared__ alignas(32) _Float16 sBf[SBF_HALFS];
  __shared__ float sbias[2 * Dm];
  int t = threadIdx.x;
  long row0 = (long)blockIdx.x * MT;
  stage_A_frag(sAf, ctx_in, row0, M, t);
  stage_B_frag(sBf, Ww, t);
  for (int i = t; i < 2 * Dm; i += BLK) sbias[i] = (i < Dm) ? Wb[i] : Ab[i - Dm];
  __syncthreads();

  int wave = t >> 5, lane = t & 31;
  v8f accP[8] = {};
  wave_gemm(sAf, sBf, wave, lane, accP);
  __syncthreads();
  stage_B_frag(sBf, Aw, t);          // reuse LDS for second weight matrix
  __syncthreads();
  v8f accQ[8] = {};
  wave_gemm(sAf, sBf, wave, lane, accQ);

  float rv = r_all[ridx];
  int half = lane >> 4, ln = lane & 15;
#pragma unroll
  for (int nt = 0; nt < 8; ++nt) {
    int n = nt * 16 + ln;
    float bW = sbias[n], bA = sbias[Dm + n];
#pragma unroll
    for (int v = 0; v < 8; ++v) {
      long gr = row0 + wave * 16 + half * 8 + v;   // C layout: M = v + 8*half
      if (gr < (long)M) {
        float P = accP[nt][v] + bW;
        float Q = scale * accQ[nt][v] + bA;
        ctx_out[gr * Dm + n] = rv * P + (1.f - rv) * fmaxf(Q, 0.f);
      }
    }
  }
}

// ---- generic GEMM: out[m*ldo + n] = A@W^T + bias --------------------------
__global__ __launch_bounds__(BLK) void gemm_kernel(
    const float* __restrict__ A, const float* __restrict__ W,
    const float* __restrict__ bias, float* __restrict__ out, int M, int ldo) {
  __shared__ alignas(32) _Float16 sAf[SAF_HALFS];
  __shared__ alignas(32) _Float16 sBf[SBF_HALFS];
  __shared__ float sbias[Dm];
  int t = threadIdx.x;
  long row0 = (long)blockIdx.x * MT;
  stage_A_frag(sAf, A, row0, M, t);
  stage_B_frag(sBf, W, t);
  for (int i = t; i < Dm; i += BLK) sbias[i] = bias[i];
  __syncthreads();
  int wave = t >> 5, lane = t & 31;
  v8f acc[8] = {};
  wave_gemm(sAf, sBf, wave, lane, acc);
  int half = lane >> 4, ln = lane & 15;
#pragma unroll
  for (int nt = 0; nt < 8; ++nt) {
    int n = nt * 16 + ln;
    float bv = sbias[n];
#pragma unroll
    for (int v = 0; v < 8; ++v) {
      long gr = row0 + wave * 16 + half * 8 + v;
      if (gr < (long)M) out[gr * (long)ldo + n] = acc[nt][v] + bv;
    }
  }
}

// ---- pointer: u[row] (+)= sum_n v[n]*tanh(qq[b,n] + (ref@Wref^T + bref)[row,n])
__global__ __launch_bounds__(BLK) void pointer_kernel(
    const float* __restrict__ ref, const float* __restrict__ Wref,
    const float* __restrict__ bref, const float* __restrict__ qq,
    const float* __restrict__ vvec, float* __restrict__ u, int M, int add) {
  __shared__ alignas(32) _Float16 sAf[SAF_HALFS];
  __shared__ alignas(32) _Float16 sBf[SBF_HALFS];
  __shared__ float sbr[Dm];
  __shared__ float sv[Dm];
  int t = threadIdx.x;
  long row0 = (long)blockIdx.x * MT;
  stage_A_frag(sAf, ref, row0, M, t);
  stage_B_frag(sBf, Wref, t);
  for (int i = t; i < Dm; i += BLK) { sbr[i] = bref[i]; sv[i] = vvec[i]; }
  __syncthreads();
  int wave = t >> 5, lane = t & 31;
  v8f acc[8] = {};
  wave_gemm(sAf, sBf, wave, lane, acc);
  int half = lane >> 4, ln = lane & 15;
#pragma unroll
  for (int v = 0; v < 8; ++v) {
    long gr = row0 + wave * 16 + half * 8 + v;
    float partial = 0.f;
    if (gr < (long)M) {
      long b = gr / S_LEN;
#pragma unroll
      for (int nt = 0; nt < 8; ++nt) {
        int n = nt * 16 + ln;
        float rr = acc[nt][v] + sbr[n];
        partial += sv[n] * tanhf(qq[b * Dm + n] + rr);
      }
    }
    // butterfly over the 16 lanes holding the same row (same half)
#pragma unroll
    for (int off = 1; off < 16; off <<= 1)
      partial += __shfl_xor(partial, off, 32);
    if (ln == 0 && gr < (long)M) {
      if (add) u[gr] += partial; else u[gr] = partial;
    }
  }
}

// ---- embeddings -----------------------------------------------------------
__global__ void embed_ctx_kernel(const float* __restrict__ X_all, const float* __restrict__ x,
                                 const float* __restrict__ w, const float* __restrict__ b,
                                 float* __restrict__ ctx) {
  long gid = (long)blockIdx.x * blockDim.x + threadIdx.x;
  if (gid >= (long)M_BIG * Dm) return;
  long row = gid >> 7; int d = (int)(gid & 127);
  long bb = row / S_LEN;
  float dx = X_all[row * 2 + 0] - x[bb * 2 + 0];
  float dy = X_all[row * 2 + 1] - x[bb * 2 + 1];
  float dist = sqrtf(dx * dx + dy * dy);
  ctx[gid] = w[d * 3 + 0] * dist + w[d * 3 + 1] * dx + w[d * 3 + 2] * dy + b[d];
}

__global__ void xe_kernel(const float* __restrict__ x, const float* __restrict__ w,
                          const float* __restrict__ b, float* __restrict__ xe) {
  int gid = blockIdx.x * blockDim.x + threadIdx.x;
  if (gid >= B_SZ * Dm) return;
  int bb = gid >> 7, d = gid & 127;
  xe[gid] = w[d * 2 + 0] * x[bb * 2 + 0] + w[d * 2 + 1] * x[bb * 2 + 1] + b[d];
}

// ---- LSTM elementwise -----------------------------------------------------
__device__ __forceinline__ float sigmf(float x) { return 1.f / (1.f + expf(-x)); }

__global__ void lstm_cell1_kernel(const float* __restrict__ gx, const float* __restrict__ gh,
                                  const float* __restrict__ gci, const float* __restrict__ gcf,
                                  const float* __restrict__ c, float* __restrict__ c_new) {
  int gid = blockIdx.x * blockDim.x + threadIdx.x;
  if (gid >= B_SZ * Dm) return;
  int bb = gid >> 7, d = gid & 127;
  const float* gxb = gx + bb * 512;
  const float* ghb = gh + bb * 512;
  float ig = sigmf(gxb[d]       + ghb[d]       + gci[gid]);
  float fg = sigmf(gxb[128 + d] + ghb[128 + d] + gcf[gid]);
  float g  = tanhf(gxb[256 + d] + ghb[256 + d]);
  c_new[gid] = fg * c[gid] + ig * g;
}

__global__ void lstm_cell2_kernel(const float* __restrict__ gx, const float* __restrict__ gh,
                                  const float* __restrict__ gco, const float* __restrict__ c_new,
                                  float* __restrict__ h_new) {
  int gid = blockIdx.x * blockDim.x + threadIdx.x;
  if (gid >= B_SZ * Dm) return;
  int bb = gid >> 7, d = gid & 127;
  float og = sigmf(gx[bb * 512 + 384 + d] + gh[bb * 512 + 384 + d] + gco[gid]);
  h_new[gid] = og * tanhf(c_new[gid]);
}

// ---- softmax over S per batch row ----------------------------------------
__global__ __launch_bounds__(256) void softmax_kernel(
    const float* __restrict__ u, const float* __restrict__ mask,
    float* __restrict__ probs, float* __restrict__ latent) {
  __shared__ float red[256];
  int b = blockIdx.x, t = threadIdx.x;
  float vals[4];
  float lmax = -3.4e38f;
#pragma unroll
  for (int i = 0; i < 4; ++i) {
    int s = t + i * 256;
    if (s < S_LEN) {
      float uv = u[b * S_LEN + s];
      latent[b * S_LEN + s] = uv;
      float xv = 100.f * tanhf(uv) + mask[b * S_LEN + s];
      vals[i] = xv;
      lmax = fmaxf(lmax, xv);
    } else vals[i] = -3.4e38f;
  }
  red[t] = lmax; __syncthreads();
  for (int o = 128; o > 0; o >>= 1) { if (t < o) red[t] = fmaxf(red[t], red[t + o]); __syncthreads(); }
  float m = red[0]; __syncthreads();
  float lsum = 0.f;
#pragma unroll
  for (int i = 0; i < 4; ++i) {
    int s = t + i * 256;
    if (s < S_LEN) { vals[i] = expf(vals[i] - m); lsum += vals[i]; }
  }
  red[t] = lsum; __syncthreads();
  for (int o = 128; o > 0; o >>= 1) { if (t < o) red[t] += red[t + o]; __syncthreads(); }
  float inv = 1.f / red[0];
#pragma unroll
  for (int i = 0; i < 4; ++i) {
    int s = t + i * 256;
    if (s < S_LEN) probs[b * S_LEN + s] = vals[i] * inv;
  }
}

extern "C" void kernel_launch(void* const* d_in, const int* in_sizes, int n_in,
                              void* d_out, int out_size, void* d_ws, size_t ws_size,
                              hipStream_t stream) {
  const float* Transcontext = (const float*)d_in[0];
  const float* x            = (const float*)d_in[1];
  const float* X_all        = (const float*)d_in[2];
  const float* mask         = (const float*)d_in[3];
  const float* h            = (const float*)d_in[4];
  const float* c            = (const float*)d_in[5];
  const float* emb_x_w      = (const float*)d_in[6];
  const float* emb_x_b      = (const float*)d_in[7];
  const float* emb_all2_w   = (const float*)d_in[8];
  const float* emb_all2_b   = (const float*)d_in[9];
  const float* W_w          = (const float*)d_in[10];
  const float* W_b          = (const float*)d_in[11];
  const float* agg_w        = (const float*)d_in[12];
  const float* agg_b        = (const float*)d_in[13];
  const float* r            = (const float*)d_in[14];
  const float* lstm_Wx      = (const float*)d_in[15];
  const float* lstm_bx      = (const float*)d_in[16];
  const float* lstm_Wh      = (const float*)d_in[17];
  const float* lstm_bh      = (const float*)d_in[18];
  const float* lstm_Wc      = (const float*)d_in[19];
  const float* lstm_bc      = (const float*)d_in[20];
  const float* ptr_v        = (const float*)d_in[21];
  const float* ptr_Wq_w     = (const float*)d_in[22];
  const float* ptr_Wq_b     = (const float*)d_in[23];
  const float* ptr_Wref_w   = (const float*)d_in[24];
  const float* ptr_Wref_b   = (const float*)d_in[25];
  const float* tptr_v       = (const float*)d_in[26];
  const float* tptr_Wq_w    = (const float*)d_in[27];
  const float* tptr_Wq_b    = (const float*)d_in[28];
  const float* tptr_Wref_w  = (const float*)d_in[29];
  const float* tptr_Wref_b  = (const float*)d_in[30];

  float* out       = (float*)d_out;
  float* out_tc    = out;                 // 65,536,000 floats
  float* out_probs = out + 65536000;      // 512,000
  float* out_h     = out + 66048000;      // 65,536
  float* out_c     = out + 66113536;      // 65,536
  float* out_lat   = out + 66179072;      // 512,000

  float* ws  = (float*)d_ws;
  float* gx  = ws;                        // 512*4*128 = 262144
  float* gh  = ws + 262144;               // 262144
  float* gci = ws + 524288;               // 65536
  float* gcf = ws + 589824;               // 65536
  float* gco = ws + 655360;               // 65536
  float* xe  = ws + 720896;               // 65536
  float* qq1 = ws + 786432;               // 65536
  float* qq2 = ws + 851968;               // 65536
  float* u   = ws + 917504;               // 512000

  float* ctx = out_tc;  // use Transcontext output slot as ctx scratch; copied over at the end

  const int nbig   = (M_BIG + MT - 1) / MT;  // 5334
  const int nsmall = (B_SZ + MT - 1) / MT;   // 6
  const float scale = 1.0f / (float)(S_LEN - 1);

  embed_ctx_kernel<<<(M_BIG * Dm + 255) / 256, 256, 0, stream>>>(X_all, x, emb_all2_w, emb_all2_b, ctx);
  xe_kernel<<<(B_SZ * Dm + 255) / 256, 256, 0, stream>>>(x, emb_x_w, emb_x_b, xe);

  for (int i = 0; i < 3; ++i)
    layer_kernel<<<nbig, BLK, 0, stream>>>(ctx, ctx, W_w + i * Dm * Dm, W_b + i * Dm,
                                           agg_w + i * Dm * Dm, agg_b + i * Dm,
                                           r, i, M_BIG, scale);

  for (int e = 0; e < 4; ++e) {
    gemm_kernel<<<nsmall, BLK, 0, stream>>>(xe, lstm_Wx + e * Dm * Dm, lstm_bx + e * Dm, gx + e * Dm, B_SZ, 512);
    gemm_kernel<<<nsmall, BLK, 0, stream>>>(h,  lstm_Wh + e * Dm * Dm, lstm_bh + e * Dm, gh + e * Dm, B_SZ, 512);
  }
  gemm_kernel<<<nsmall, BLK, 0, stream>>>(c, lstm_Wc + 0 * Dm * Dm, lstm_bc + 0 * Dm, gci, B_SZ, Dm);
  gemm_kernel<<<nsmall, BLK, 0, stream>>>(c, lstm_Wc + 1 * Dm * Dm, lstm_bc + 1 * Dm, gcf, B_SZ, Dm);
  lstm_cell1_kernel<<<(B_SZ * Dm + 255) / 256, 256, 0, stream>>>(gx, gh, gci, gcf, c, out_c);
  gemm_kernel<<<nsmall, BLK, 0, stream>>>(out_c, lstm_Wc + 2 * Dm * Dm, lstm_bc + 2 * Dm, gco, B_SZ, Dm);
  lstm_cell2_kernel<<<(B_SZ * Dm + 255) / 256, 256, 0, stream>>>(gx, gh, gco, out_c, out_h);

  gemm_kernel<<<nsmall, BLK, 0, stream>>>(out_h, ptr_Wq_w,  ptr_Wq_b,  qq1, B_SZ, Dm);
  gemm_kernel<<<nsmall, BLK, 0, stream>>>(out_h, tptr_Wq_w, tptr_Wq_b, qq2, B_SZ, Dm);

  pointer_kernel<<<nbig, BLK, 0, stream>>>(ctx, ptr_Wref_w, ptr_Wref_b, qq1, ptr_v, u, M_BIG, 0);
  pointer_kernel<<<nbig, BLK, 0, stream>>>(Transcontext, tptr_Wref_w, tptr_Wref_b, qq2, tptr_v, u, M_BIG, 1);

  softmax_kernel<<<B_SZ, 256, 0, stream>>>(u, mask, out_probs, out_lat);

  // ctx scratch is no longer needed: emit the passthrough Transcontext output last.
  hipMemcpyAsync(out_tc, Transcontext, (size_t)M_BIG * Dm * sizeof(float),
                 hipMemcpyDeviceToDevice, stream);
}
